// NODE_DMD_24541443129468
// MI455X (gfx1250) — matrix operand order repaired and checked by reading, not compiled, not run on hardware
//
#include <hip/hip_runtime.h>
#include <hip/hip_bf16.h>

#define N_PTS 200000
#define HID 256
#define R16 16
#define TWO_R 32
#define ODE_STEPS 20
#define WAVES_PER_BLOCK 8

typedef _Float16 h2_t  __attribute__((ext_vector_type(2)));
typedef _Float16 h8_t  __attribute__((ext_vector_type(8)));
typedef _Float16 v16h  __attribute__((ext_vector_type(16)));
typedef float    v8f   __attribute__((ext_vector_type(8)));
typedef unsigned int u32x4 __attribute__((ext_vector_type(4)));
typedef int          i32x4 __attribute__((ext_vector_type(4)));
typedef int          i32x8 __attribute__((ext_vector_type(8)));

// Workspace half-region layout (element offsets):
//   We1T : [256][32]   -> 8192     (K padded 4->32)
//   We2T : [256][256]  -> 65536
//   Wd1T : [256][64]   -> 16384    (K padded 34->64)
//   Wd2T : [256][256]  -> 65536
//   Wd3T : [16][256]   -> 4096     (N padded 2->16)
#define OFF_WE1T 0
#define OFF_WE2T 8192
#define OFF_WD1T 73728
#define OFF_WD2T 90112
#define OFF_WD3T 155648
#define TOTAL_HALVES 159744

#if defined(__HIP_DEVICE_COMPILE__) && __has_builtin(__builtin_amdgcn_tensor_load_to_lds)
#define USE_TDM 1
#else
#define USE_TDM 0
#endif

__device__ __forceinline__ v16h join16(h8_t lo, h8_t hi) {
  return __builtin_shufflevector(lo, hi, 0,1,2,3,4,5,6,7,8,9,10,11,12,13,14,15);
}

// B fragment (32x16 f16: lane-group g, halves 0..7 <-> k = g*8 + i,
// halves 8..15 <-> k = 16 + g*8 + (i-8)). Weights stored transposed [n][Kpad]
// so each 8-half run is contiguous: two 16-byte LDS loads.
__device__ __forceinline__ v16h load_bfrag(const _Float16* wT, int Kpad,
                                           int nt, int kt, int lane) {
  int n = nt * 16 + (lane & 15);
  int g = lane >> 4;
  const _Float16* p = wT + n * Kpad + kt * 32 + g * 8;
  h8_t lo = *(const h8_t*)p;
  h8_t hi = *(const h8_t*)(p + 16);
  return join16(lo, hi);
}

// A fragment (16x32 f16 per ISA: lane m holds row m; halves 0..7 <-> k =
// kt*32 + g*8 + i, halves 8..15 <-> k = kt*32 + 16 + g*8 + (i-8)).
// hrow = row-major [16][HID] LDS stage, pointer already at row m.
__device__ __forceinline__ v16h load_afrag(const _Float16* hrow, int kt, int g) {
  const _Float16* p = hrow + kt * 32 + g * 8;
  h8_t lo = *(const h8_t*)p;
  h8_t hi = *(const h8_t*)(p + 16);
  return join16(lo, hi);
}

__device__ __forceinline__ v8f wmma16(v16h a, v16h b, v8f c) {
  return __builtin_amdgcn_wmma_f32_16x16x32_f16(false, a, false, b,
                                                (short)0, c, false, false);
}

__device__ __forceinline__ v8f splat8(float v) {
  v8f c;
#pragma unroll
  for (int j = 0; j < 8; ++j) c[j] = v;
  return c;
}

// Single-op f32 ReLU: v_med3_f32(x, 0, +inf).
__device__ __forceinline__ float relu1(float x) {
  return __builtin_amdgcn_fmed3f(x, 0.0f, __builtin_inff());
}

// Packed f16 max(h, 0): v_pk_max_num_f16.
__device__ __forceinline__ h2_t pk_relu(h2_t h) {
  const h2_t z = {(_Float16)0.0f, (_Float16)0.0f};
#if __has_builtin(__builtin_elementwise_maximumnum)
  return __builtin_elementwise_maximumnum(h, z);
#else
  return __builtin_elementwise_max(h, z);
#endif
}

// ReLU + f32->f16 + store one 16x16 C tile to the LDS h-stage:
// v_cvt_pk_rtz_f16_f32 (2 elems/op) then packed v_pk_max_num_f16.
// base = hrow + (g*8)*HID + col; rows advance by HID halves.
__device__ __forceinline__ void relu_store8(v8f c, _Float16* base) {
#pragma unroll
  for (int t = 0; t < 4; ++t) {
    h2_t h = __builtin_bit_cast(
        h2_t, __builtin_amdgcn_cvt_pkrtz(c[2 * t], c[2 * t + 1]));
    h = pk_relu(h);
    base[(2 * t) * HID]     = h[0];
    base[(2 * t + 1) * HID] = h[1];
  }
}

#if USE_TDM
// Raw LDS byte offset of a (dynamic) shared-memory pointer for the D#.
__device__ __forceinline__ unsigned lds_offset_of(const void* p) {
  return (unsigned)(unsigned long long)
      (const __attribute__((address_space(3))) void*)p;
}

// One-shot TDM: contiguous global -> LDS, n8 elements of 8 bytes (1-D tile).
// D# per ISA 8.3/8.4: group0 = {count=1, lds_addr, global_addr, type=2},
// group1 = {data_size=3, tensor_dim0 = tile_dim0 = n8, stride0 = n8}.
__device__ __forceinline__ void tdm_load_to_lds(const void* gsrc,
                                                unsigned lds_addr,
                                                unsigned n8) {
  unsigned long long ga = (unsigned long long)(__SIZE_TYPE__)gsrc;
  u32x4 g0;
  g0[0] = 1u;                                       // count=1, user desc
  g0[1] = lds_addr;                                 // lds_addr[31:0]
  g0[2] = (unsigned)(ga & 0xFFFFFFFFu);             // global_addr[31:0]
  g0[3] = (unsigned)((ga >> 32) & 0x01FFFFFFu) | 0x80000000u; // [56:32]|type=2
  i32x8 g1;
  g1[0] = 0x30000;                                  // data_size=3 (8B), mask=0
  g1[1] = (int)((n8 & 0xFFFFu) << 16);              // tensor_dim0[15:0]
  g1[2] = (int)((n8 >> 16) & 0xFFFFu) | (1 << 16);  // dim0[31:16] | dim1=1
  g1[3] = (int)((n8 & 0xFFFFu) << 16);              // tile_dim0
  g1[4] = 0;                                        // tile_dim1/2 = 0 (1-D)
  g1[5] = (int)n8;                                  // dim0_stride[31:0]
  g1[6] = 0;
  g1[7] = 0;
  i32x4 z4;
#pragma unroll
  for (int i = 0; i < 4; ++i) z4[i] = 0;
#if defined(__clang_major__) && (__clang_major__ >= 23)
  i32x8 z8;
#pragma unroll
  for (int i = 0; i < 8; ++i) z8[i] = 0;
  __builtin_amdgcn_tensor_load_to_lds(g0, g1, z4, z4, z8, 0);
#else
  __builtin_amdgcn_tensor_load_to_lds(g0, g1, z4, z4, 0);
#endif
}

__device__ __forceinline__ void tdm_wait() {
#if __has_builtin(__builtin_amdgcn_s_wait_tensorcnt)
  __builtin_amdgcn_s_wait_tensorcnt(0);
#else
  asm volatile("s_wait_tensorcnt 0x0" ::: "memory");
#endif
}
#endif  // USE_TDM

// Stage `halves` f16 weights from global `src` into LDS `dst`.
__device__ __forceinline__ void stage_weights(const _Float16* __restrict__ src,
                                              _Float16* dst, int halves,
                                              int tid) {
#if USE_TDM
  if (tid == 0) {
    tdm_load_to_lds(src, lds_offset_of(dst), (unsigned)(halves / 4));
    tdm_wait();
  }
#else
  const h8_t* s = (const h8_t*)src;
  h8_t* d = (h8_t*)dst;
  for (int i = tid; i < halves / 8; i += 256) d[i] = s[i];
#endif
}

// ---------------------------------------------------------------- prep ----
__global__ __launch_bounds__(256) void prep_kernel(
    const float* __restrict__ We1, const float* __restrict__ We2,
    const float* __restrict__ Wd1, const float* __restrict__ Wd2,
    const float* __restrict__ Wd3, float* __restrict__ pool_acc,
    _Float16* __restrict__ wT) {
  int idx = blockIdx.x * 256 + threadIdx.x;
  if (idx < 256) pool_acc[idx] = 0.0f;
  if (idx < OFF_WE2T) {                       // We1T, K padded to 32
    int n = idx / 32, k = idx % 32;
    wT[idx] = (_Float16)(k < 4 ? We1[k * 256 + n] : 0.0f);
  } else if (idx < OFF_WD1T) {                // We2T
    int e = idx - OFF_WE2T; int n = e / 256, k = e % 256;
    wT[idx] = (_Float16)We2[k * 256 + n];
  } else if (idx < OFF_WD2T) {                // Wd1T, K padded to 64
    int e = idx - OFF_WD1T; int n = e / 64, k = e % 64;
    wT[idx] = (_Float16)(k < 34 ? Wd1[k * 256 + n] : 0.0f);
  } else if (idx < OFF_WD3T) {                // Wd2T
    int e = idx - OFF_WD2T; int n = e / 256, k = e % 256;
    wT[idx] = (_Float16)Wd2[k * 256 + n];
  } else if (idx < TOTAL_HALVES) {            // Wd3T, N padded to 16
    int e = idx - OFF_WD3T; int n = e / 256, k = e % 256;
    wT[idx] = (_Float16)(n < 2 ? Wd3[k * 2 + n] : 0.0f);
  }
}

// ------------------------------------------------------------- encoder ----
__global__ __launch_bounds__(256) void encoder_kernel(
    const float* __restrict__ coords, const float* __restrict__ yprev,
    const float* __restrict__ b1, const float* __restrict__ b2,
    const _Float16* __restrict__ wT, float* __restrict__ pool_acc) {
  extern __shared__ char smem[];
  _Float16* we1T   = (_Float16*)smem;                 // 8192 halves
  _Float16* we2T   = we1T + 8192;                     // 65536 halves
  _Float16* hstage = we2T + 65536;                    // 8 waves * 16*256
  float*    pool_l = (float*)(hstage + WAVES_PER_BLOCK * 16 * HID);

  int tid = threadIdx.x;
  stage_weights(wT, we1T, OFF_WD1T, tid);   // We1T + We2T contiguous
  pool_l[tid] = 0.0f;
  __syncthreads();

  int wave = tid >> 5, lane = tid & 31, m = lane & 15, g = lane >> 4;
  long tileBase = ((long)blockIdx.x * WAVES_PER_BLOCK + wave) * 16;
  _Float16* hrow = hstage + wave * 16 * HID;

  // ---- layer 1: [16,4pad32] x [32,256] ----
  v16h a1;
#pragma unroll
  for (int i = 0; i < 16; ++i) a1[i] = (_Float16)0.0f;
  long row = tileBase + m;
  if (g == 0 && row < N_PTS) {     // k=0..3 live in halves 0..3 of group 0
    a1[0] = (_Float16)coords[2 * row];
    a1[1] = (_Float16)coords[2 * row + 1];
    a1[2] = (_Float16)yprev[2 * row];
    a1[3] = (_Float16)yprev[2 * row + 1];
  }
#pragma unroll 4
  for (int nt = 0; nt < 16; ++nt) {
    v8f c = splat8(b1[nt * 16 + m]);
    c = wmma16(a1, load_bfrag(we1T, 32, nt, 0, lane), c);
    relu_store8(c, hrow + (g * 8) * HID + nt * 16 + m);
  }

  // ---- layer 2: [16,256] x [256,256] + masked mean-pool partials ----
  v16h af[8];
#pragma unroll
  for (int kt = 0; kt < 8; ++kt) af[kt] = load_afrag(hrow + m * HID, kt, g);
#pragma unroll 4
  for (int nt = 0; nt < 16; ++nt) {
    v8f c = splat8(b2[nt * 16 + m]);
#pragma unroll
    for (int kt = 0; kt < 8; ++kt)
      c = wmma16(af[kt], load_bfrag(we2T, 256, nt, kt, lane), c);
    float part = 0.0f;
#pragma unroll
    for (int j = 0; j < 8; ++j) {
      if (tileBase + g * 8 + j < N_PTS) part += relu1(c[j]);  // mask pad rows
    }
    atomicAdd(&pool_l[nt * 16 + m], part);
  }
  __syncthreads();
  atomicAdd(&pool_acc[tid], pool_l[tid]);
}

// --------------------------------------------------- pooled head + ODE ----
__global__ __launch_bounds__(256) void ode_kernel(
    const float* __restrict__ pool_acc,
    const float* __restrict__ Wpool, const float* __restrict__ bpool,
    const float* __restrict__ Wmu,  const float* __restrict__ bmu,
    const float* __restrict__ Wlv,  const float* __restrict__ blv,
    const float* __restrict__ Wlam, const float* __restrict__ blam,
    const float* __restrict__ eps,  const float* __restrict__ t_next,
    const float* __restrict__ Wo1,  const float* __restrict__ bo1,
    const float* __restrict__ Wo2,  const float* __restrict__ bo2,
    const float* __restrict__ Wo3,  const float* __restrict__ bo3,
    float* __restrict__ phi_ws, float* __restrict__ out_tail) {
  __shared__ float mean_l[HID], pooled_l[HID], z_l[4 * R16 + 1], h_l[HID];
  __shared__ float mu_l[TWO_R], lv_l[TWO_R], lam_l[TWO_R], phi_l[TWO_R];
  int t = threadIdx.x;
  mean_l[t] = pool_acc[t] * (1.0f / (float)N_PTS);
  __syncthreads();
  float s = bpool[t];
  for (int k = 0; k < HID; ++k) s += mean_l[k] * Wpool[k * HID + t];
  pooled_l[t] = relu1(s);
  __syncthreads();
  if (t < 96) {
    int which = t >> 5, i = t & 31;
    const float* W = which == 0 ? Wmu : (which == 1 ? Wlv : Wlam);
    const float* b = which == 0 ? bmu : (which == 1 ? blv : blam);
    float v = b[i];
    for (int k = 0; k < HID; ++k) v += pooled_l[k] * W[k * TWO_R + i];
    out_tail[which * TWO_R + i] = v;          // mu | logvar | lambda
    if (which == 0) mu_l[i] = v;
    else if (which == 1) lv_l[i] = v;
    else lam_l[i] = v;
  }
  __syncthreads();
  if (t < TWO_R) phi_l[t] = mu_l[t] + eps[t] * __expf(0.5f * lv_l[t]);
  __syncthreads();
  float dt = t_next[0] * (1.0f / (float)ODE_STEPS);
  for (int step = 0; step < ODE_STEPS; ++step) {
    if (t < TWO_R) { z_l[t] = phi_l[t]; z_l[TWO_R + t] = lam_l[t]; }
    if (t == 0) z_l[4 * R16] = (float)step * dt;
    __syncthreads();
    float a = bo1[t];
    for (int k = 0; k < 4 * R16 + 1; ++k) a += z_l[k] * Wo1[k * HID + t];
    a = relu1(a);
    __syncthreads();
    h_l[t] = a;
    __syncthreads();
    float b2v = bo2[t];
    for (int k = 0; k < HID; ++k) b2v += h_l[k] * Wo2[k * HID + t];
    b2v = relu1(b2v);
    __syncthreads();
    h_l[t] = b2v;
    __syncthreads();
    if (t < TWO_R) {
      float d = bo3[t];
      for (int k = 0; k < HID; ++k) d += h_l[k] * Wo3[k * TWO_R + t];
      phi_l[t] += dt * d;
    }
    __syncthreads();
  }
  if (t < TWO_R) phi_ws[t] = phi_l[t];
}

// ------------------------------------------------------------- decoder ----
__global__ __launch_bounds__(256) void decoder_kernel(
    const float* __restrict__ coords,
    const float* __restrict__ bd1, const float* __restrict__ bd2,
    const float* __restrict__ bd3,
    const _Float16* __restrict__ wTdec,   // wT + OFF_WD1T
    const float* __restrict__ phi_ws, float* __restrict__ u_out) {
  extern __shared__ char smem[];
  _Float16* wd1T   = (_Float16*)smem;                 // 16384 halves
  _Float16* wd2T   = wd1T + 16384;                    // 65536 halves
  _Float16* wd3T   = wd2T + 65536;                    // 4096 halves
  _Float16* hstage = wd3T + 4096;                     // 8 * 16*256
  float*    phi_l  = (float*)(hstage + WAVES_PER_BLOCK * 16 * HID);

  int tid = threadIdx.x;
  stage_weights(wTdec, wd1T, TOTAL_HALVES - OFF_WD1T, tid);
  if (tid < TWO_R) phi_l[tid] = phi_ws[tid];
  __syncthreads();

  int wave = tid >> 5, lane = tid & 31, m = lane & 15, g = lane >> 4;
  long tileBase = ((long)blockIdx.x * WAVES_PER_BLOCK + wave) * 16;
  _Float16* hrow = hstage + wave * 16 * HID;
  long row = tileBase + m;
  bool valid = row < N_PTS;

  // ---- layer 1: feat = [coords(2) | phi(32)] padded to K=64 ----
  v16h a0, a1;
#pragma unroll
  for (int i = 0; i < 16; ++i) {
    int kk = (i < 8) ? (g * 8 + i) : (16 + g * 8 + (i - 8));
    float f0, f1;
    // kt = 0 half (k = kk)
    if (kk < 2)       f0 = valid ? coords[2 * row + kk] : 0.0f;
    else if (kk < 34) f0 = phi_l[kk - 2];
    else              f0 = 0.0f;
    // kt = 1 half (k = 32 + kk)
    int k1 = 32 + kk;
    f1 = (k1 < 34) ? phi_l[k1 - 2] : 0.0f;
    a0[i] = (_Float16)f0;
    a1[i] = (_Float16)f1;
  }
#pragma unroll 4
  for (int nt = 0; nt < 16; ++nt) {
    v8f c = splat8(bd1[nt * 16 + m]);
    c = wmma16(a0, load_bfrag(wd1T, 64, nt, 0, lane), c);
    c = wmma16(a1, load_bfrag(wd1T, 64, nt, 1, lane), c);
    relu_store8(c, hrow + (g * 8) * HID + nt * 16 + m);
  }

  // ---- layer 2: [16,256] x [256,256] ----
  v16h af[8];
#pragma unroll
  for (int kt = 0; kt < 8; ++kt) af[kt] = load_afrag(hrow + m * HID, kt, g);
#pragma unroll 4
  for (int nt = 0; nt < 16; ++nt) {
    v8f c = splat8(bd2[nt * 16 + m]);
#pragma unroll
    for (int kt = 0; kt < 8; ++kt)
      c = wmma16(af[kt], load_bfrag(wd2T, 256, nt, kt, lane), c);
    relu_store8(c, hrow + (g * 8) * HID + nt * 16 + m);
  }

  // ---- layer 3: [16,256] x [256,2pad16] ----
#pragma unroll
  for (int kt = 0; kt < 8; ++kt) af[kt] = load_afrag(hrow + m * HID, kt, g);
  {
    v8f c = splat8(m < 2 ? bd3[m] : 0.0f);
#pragma unroll
    for (int kt = 0; kt < 8; ++kt)
      c = wmma16(af[kt], load_bfrag(wd3T, 256, 0, kt, lane), c);
    if (m < 2) {
#pragma unroll
      for (int j = 0; j < 8; ++j) {
        long r = tileBase + g * 8 + j;
        if (r < N_PTS) u_out[r * 2 + m] = c[j];
      }
    }
  }
}

// -------------------------------------------------------------- launch ----
extern "C" void kernel_launch(void* const* d_in, const int* in_sizes, int n_in,
                              void* d_out, int out_size, void* d_ws, size_t ws_size,
                              hipStream_t stream) {
  const float* coords = (const float*)d_in[0];
  const float* yprev  = (const float*)d_in[1];
  /* d_in[2] = t_prev (unused, reference integrates from 0) */
  const float* t_next = (const float*)d_in[3];
  const float* eps    = (const float*)d_in[4];
  const float* We1 = (const float*)d_in[5];  const float* be1 = (const float*)d_in[6];
  const float* We2 = (const float*)d_in[7];  const float* be2 = (const float*)d_in[8];
  const float* Wpl = (const float*)d_in[9];  const float* bpl = (const float*)d_in[10];
  const float* Wmu = (const float*)d_in[11]; const float* bmu = (const float*)d_in[12];
  const float* Wlv = (const float*)d_in[13]; const float* blv = (const float*)d_in[14];
  const float* Wlm = (const float*)d_in[15]; const float* blm = (const float*)d_in[16];
  const float* Wo1 = (const float*)d_in[17]; const float* bo1 = (const float*)d_in[18];
  const float* Wo2 = (const float*)d_in[19]; const float* bo2 = (const float*)d_in[20];
  const float* Wo3 = (const float*)d_in[21]; const float* bo3 = (const float*)d_in[22];
  const float* Wd1 = (const float*)d_in[23]; const float* bd1 = (const float*)d_in[24];
  const float* Wd2 = (const float*)d_in[25]; const float* bd2 = (const float*)d_in[26];
  const float* Wd3 = (const float*)d_in[27]; const float* bd3 = (const float*)d_in[28];

  float*     pool_acc = (float*)d_ws;
  float*     phi_ws   = (float*)((char*)d_ws + 1024);
  _Float16*  wT       = (_Float16*)((char*)d_ws + 2048);
  float*     out      = (float*)d_out;

  // 1) weights -> f16 transposed/padded, zero pool accumulator
  prep_kernel<<<(TOTAL_HALVES + 255) / 256, 256, 0, stream>>>(
      We1, We2, Wd1, Wd2, Wd3, pool_acc, wT);

  // 2) encoder + mean-pool partials
  int tiles = (N_PTS + 15) / 16;                       // 12500
  int blocks = (tiles + WAVES_PER_BLOCK - 1) / WAVES_PER_BLOCK;  // 1563
  size_t smem_enc = (size_t)(OFF_WD1T + WAVES_PER_BLOCK * 16 * HID) * 2 + 256 * 4;
  encoder_kernel<<<blocks, 256, smem_enc, stream>>>(
      coords, yprev, be1, be2, wT, pool_acc);

  // 3) pooled head + 20 Euler steps (tiny, serial)
  ode_kernel<<<1, 256, 0, stream>>>(
      pool_acc, Wpl, bpl, Wmu, bmu, Wlv, blv, Wlm, blm, eps, t_next,
      Wo1, bo1, Wo2, bo2, Wo3, bo3, phi_ws, out + (size_t)N_PTS * 2);

  // 4) decoder -> u_pred
  size_t smem_dec = (size_t)((TOTAL_HALVES - OFF_WD1T) +
                             WAVES_PER_BLOCK * 16 * HID) * 2 + TWO_R * 4;
  decoder_kernel<<<blocks, 256, smem_dec, stream>>>(
      coords, bd1, bd2, bd3, wT + OFF_WD1T, phi_ws, out);
}
